// LNN_71296457114062
// MI455X (gfx1250) — compile-verified, matrix operand order checked by reading, not run on hardware
//
#include <hip/hip_runtime.h>
#include <hip/hip_bf16.h>

typedef __attribute__((ext_vector_type(16))) __bf16 v16bf;
typedef __attribute__((ext_vector_type(8)))  __bf16 v8bf;
typedef __attribute__((ext_vector_type(8)))  float  v8f;
typedef __attribute__((ext_vector_type(4)))  float  v4f;

#define WMMA_BF16(a, b, c) \
  __builtin_amdgcn_wmma_f32_16x16x32_bf16(false, (a), false, (b), (short)0, (c), false, false)

// Hardware tanh (CDNA5 V_TANH_F32 trans op)
__device__ __forceinline__ float fast_tanh(float x) {
#if __has_builtin(__builtin_amdgcn_tanhf)
  return __builtin_amdgcn_tanhf(x);
#else
  float r;
  asm volatile("v_tanh_f32 %0, %1" : "=v"(r) : "v"(x));
  return r;
#endif
}
// sigmoid via tanh: avoids exp + divide on the serial critical path
__device__ __forceinline__ float fast_sigmoid(float x) {
  return 0.5f * (1.0f + fast_tanh(0.5f * x));
}

// A/B fragment (16x32 bf16, wave32). Per CDNA5 ISA layout:
// lane L holds row (L&15); halves: lanes 0-15 -> K {k0+0..7, k0+16..23},
// lanes 16-31 -> K {k0+8..15, k0+24..31}. Caller passes p0 = row_start + k0 + koff.
__device__ __forceinline__ v16bf frag_from_bf16(const __bf16* p0) {
  v8bf lo = *(const v8bf*)p0;        // 16B -> b128 (ds_load_b128 / global_load_b128)
  v8bf hi = *(const v8bf*)(p0 + 16);
  v16bf f;
#pragma unroll
  for (int i = 0; i < 8; ++i) { f[i] = lo[i]; f[8 + i] = hi[i]; }
  return f;
}

__device__ __forceinline__ float gelu_exact(float v) {
  return 0.5f * v * (1.0f + erff(v * 0.70710678118654752f));
}

// ---- conversion kernels: fp32 -> bf16, 8 elems/thread, b128 traffic ---------
__global__ void __launch_bounds__(256) k_cvt_f32_bf16(
    const float* __restrict__ in, __bf16* __restrict__ out, int n8) {
  int i = blockIdx.x * blockDim.x + threadIdx.x;
  if (i >= n8) return;
  const float* p = in + (size_t)i * 8;
  v8bf o;
#pragma unroll
  for (int j = 0; j < 8; ++j) o[j] = (__bf16)p[j];
  *(v8bf*)(out + (size_t)i * 8) = o;
}

// Wb[:, :512] (ld 544) -> packed bf16 [128,512]
__global__ void __launch_bounds__(256) k_cvt_wbx(
    const float* __restrict__ Wb, __bf16* __restrict__ out) {
  int i = blockIdx.x * blockDim.x + threadIdx.x;  // 8192 chunks of 8
  int n = i >> 6;
  int c8 = (i & 63) << 3;
  const float* p = Wb + n * 544 + c8;
  v8bf o;
#pragma unroll
  for (int j = 0; j < 8; ++j) o[j] = (__bf16)p[j];
  *(v8bf*)(out + (size_t)n * 512 + c8) = o;
}

// ---- kernel 1: h0 = GELU(xb @ W0b^T + b0) bf16 [65536, 512] -----------------
// block: 8 waves, tile M=256 x N=128; B staged in 64KB LDS chunks (K=256).
__global__ void __launch_bounds__(256) k_gemm1(
    const __bf16* __restrict__ A, const __bf16* __restrict__ Bw,
    const float* __restrict__ b0, __bf16* __restrict__ C) {
  extern __shared__ __bf16 Bs[];  // 128 x 256 bf16 = 64KB
  const int tid = threadIdx.x;
  const int lane = tid & 31;
  const int w = tid >> 5;
  const int n0 = (blockIdx.x & 3) * 128;
  const int m0 = (blockIdx.x >> 2) * 256 + w * 32;
  const int r = lane & 15;
  const int col = r;
  const int koff = (lane & 16) ? 8 : 0;
  const int rowOff = (lane & 16) ? 8 : 0;

  v8f acc[2][8] = {};
  for (int kc = 0; kc < 1024; kc += 256) {
    __syncthreads();
    for (int i = tid; i < 4096; i += 256) {       // 128x256 bf16 cooperative copy
      int row = i >> 5, c8 = (i & 31) << 3;
      *(v8bf*)&Bs[row * 256 + c8] =
          *(const v8bf*)&Bw[(size_t)(n0 + row) * 1024 + kc + c8];
    }
    __syncthreads();
#pragma unroll
    for (int k0 = 0; k0 < 256; k0 += 32) {
      v16bf a0 = frag_from_bf16(A + (size_t)(m0 + r) * 1024 + kc + k0 + koff);
      v16bf a1 = frag_from_bf16(A + (size_t)(m0 + 16 + r) * 1024 + kc + k0 + koff);
      v16bf bfr[8];
#pragma unroll
      for (int j = 0; j < 8; ++j)                 // issue all 16 ds_loads up front
        bfr[j] = frag_from_bf16(&Bs[(16 * j + r) * 256 + k0 + koff]);
#pragma unroll
      for (int j = 0; j < 8; ++j) {               // 16 back-to-back WMMAs
        acc[0][j] = WMMA_BF16(a0, bfr[j], acc[0][j]);
        acc[1][j] = WMMA_BF16(a1, bfr[j], acc[1][j]);
      }
    }
  }
  // epilogue: GELU, restage per-wave 32x128 tile in LDS, coalesced b128 stores
  __syncthreads();
  __bf16* St = Bs + (size_t)w * 4096;
#pragma unroll
  for (int s = 0; s < 2; ++s)
#pragma unroll
    for (int j = 0; j < 8; ++j) {
      float bias = b0[n0 + 16 * j + col];
#pragma unroll
      for (int v = 0; v < 8; ++v) {
        float val = gelu_exact(acc[s][j][v] + bias);
        St[(16 * s + rowOff + v) * 128 + 16 * j + col] = (__bf16)val;
      }
    }
#pragma unroll
  for (int i = lane; i < 512; i += 32) {          // 32x128 bf16 -> global, b128
    int row = i >> 4, c8 = (i & 15) << 3;
    *(v8bf*)&C[(size_t)(m0 + row) * 512 + n0 + c8] = *(const v8bf*)&St[row * 128 + c8];
  }
}

// ---- kernel 2: pre[m][n] = h0 @ Wbx^T + bb, fp32 [65536, 128] (m = b*512+t) -
__global__ void __launch_bounds__(256) k_gemm2(
    const __bf16* __restrict__ A, const __bf16* __restrict__ Bw,
    const float* __restrict__ bb, float* __restrict__ pre) {
  extern __shared__ __bf16 Bs[];  // 128 x 256 bf16 = 64KB (also f32 staging)
  const int tid = threadIdx.x;
  const int lane = tid & 31;
  const int w = tid >> 5;
  const int m0 = blockIdx.x * 256 + w * 32;
  const int r = lane & 15;
  const int col = r;
  const int koff = (lane & 16) ? 8 : 0;
  const int rowOff = (lane & 16) ? 8 : 0;

  v8f acc[2][8] = {};
  for (int kc = 0; kc < 512; kc += 256) {
    __syncthreads();
    for (int i = tid; i < 4096; i += 256) {
      int row = i >> 5, c8 = (i & 31) << 3;
      *(v8bf*)&Bs[row * 256 + c8] =
          *(const v8bf*)&Bw[(size_t)row * 512 + kc + c8];
    }
    __syncthreads();
#pragma unroll
    for (int k0 = 0; k0 < 256; k0 += 32) {
      v16bf a0 = frag_from_bf16(A + (size_t)(m0 + r) * 512 + kc + k0 + koff);
      v16bf a1 = frag_from_bf16(A + (size_t)(m0 + 16 + r) * 512 + kc + k0 + koff);
      v16bf bfr[8];
#pragma unroll
      for (int j = 0; j < 8; ++j)
        bfr[j] = frag_from_bf16(&Bs[(16 * j + r) * 256 + k0 + koff]);
#pragma unroll
      for (int j = 0; j < 8; ++j) {
        acc[0][j] = WMMA_BF16(a0, bfr[j], acc[0][j]);
        acc[1][j] = WMMA_BF16(a1, bfr[j], acc[1][j]);
      }
    }
  }
  // epilogue: bias add, stage 16x128 f32 per wave (64KB total), coalesced store
  __syncthreads();
  float* Sf = (float*)Bs + (size_t)w * 2048;  // 16 x 128 f32
#pragma unroll
  for (int s = 0; s < 2; ++s) {
#pragma unroll
    for (int j = 0; j < 8; ++j) {
      float bias = bb[16 * j + col];
#pragma unroll
      for (int v = 0; v < 8; ++v)
        Sf[(rowOff + v) * 128 + 16 * j + col] = acc[s][j][v] + bias;
    }
#pragma unroll
    for (int i = lane; i < 512; i += 32) {    // 16x128 f32 -> global, b128
      int row = i >> 5, c4 = (i & 31) << 2;
      *(v4f*)&pre[(size_t)(m0 + 16 * s + row) * 128 + c4] = *(v4f*)&Sf[row * 128 + c4];
    }
  }
}

// ---- kernel 3: serial CfC scan; 8 blocks x 1 wave, 16 batch rows each -------
__global__ void __launch_bounds__(32) k_scan(
    const float* __restrict__ pre, const float* __restrict__ Wb,
    const float* __restrict__ Wff1, const float* __restrict__ bff1,
    const float* __restrict__ Wff2, const float* __restrict__ bff2,
    const float* __restrict__ Wta,  const float* __restrict__ bta,
    const float* __restrict__ Wtb,  const float* __restrict__ btb,
    float* __restrict__ seq) {
  __shared__ __bf16 WhL[128 * 128];  // fused heads [n][k]: ff1|ff2|ta|tb
  __shared__ __bf16 WsL[128 * 32];   // state weights Wb[:,512:]
  __shared__ float  bhL[128];
  __shared__ float  gbuf[16 * 128];  // hG staging (D-layout -> A-layout)
  __shared__ __bf16 hbuf[16 * 32];

  const int lane = threadIdx.x;
  const int bb0 = blockIdx.x * 16;

  const float* Wsrc[4] = {Wff1, Wff2, Wta, Wtb};
  const float* bsrc[4] = {bff1, bff2, bta, btb};
  for (int i = lane; i < 128 * 128; i += 32) {
    int n = i >> 7, k = i & 127;
    WhL[i] = (__bf16)Wsrc[n >> 5][(n & 31) * 128 + k];
  }
  for (int i = lane; i < 128 * 32; i += 32) {
    int n = i >> 5, k = i & 31;
    WsL[i] = (__bf16)Wb[n * 544 + 512 + k];
  }
  for (int i = lane; i < 128; i += 32) bhL[i] = bsrc[i >> 5][i & 31];
  __syncthreads();

  const int col = lane & 15;
  const int rowOff = (lane & 16) ? 8 : 0;
  const int koff = (lane & 16) ? 8 : 0;
  const int r = lane & 15;

  // hoist loop-invariant state-weight B fragments (8 x 8 VGPRs)
  v16bf wsF[8];
#pragma unroll
  for (int j = 0; j < 8; ++j)
    wsF[j] = frag_from_bf16(&WsL[(size_t)(16 * j + col) * 32 + koff]);

  v16bf hA{};  // h state A-fragment (16x32), zero init
  for (int t = 0; t < 512; ++t) {
    // state GEMM: hG[16,128] = h @ Wbh^T, scattered to gbuf in D-layout
#pragma unroll
    for (int j = 0; j < 8; ++j) {
      v8f s{};
      s = WMMA_BF16(hA, wsF[j], s);
#pragma unroll
      for (int v = 0; v < 8; ++v) gbuf[(rowOff + v) * 128 + 16 * j + col] = s[v];
    }
    __syncthreads();

    // g = lecun_tanh(pre + hG), built directly as A-fragments (hardware tanh)
    v16bf gA[4];
#pragma unroll
    for (int f = 0; f < 4; ++f) {
      const float* gb = &gbuf[r * 128 + 32 * f + koff];
      const float* pr = pre + ((size_t)(bb0 + r) * 512 + t) * 128 + 32 * f + koff;
#pragma unroll
      for (int i = 0; i < 8; ++i) {
        gA[f][i]     = (__bf16)(1.7159f * fast_tanh(0.666f * (gb[i]      + pr[i])));
        gA[f][8 + i] = (__bf16)(1.7159f * fast_tanh(0.666f * (gb[16 + i] + pr[16 + i])));
      }
    }

    // fused heads: [16,128] @ Wheads^T -> [16,128]
    v8f acc[8];
#pragma unroll
    for (int j = 0; j < 8; ++j) {
      v16bf bfr[4];
#pragma unroll
      for (int f = 0; f < 4; ++f)
        bfr[f] = frag_from_bf16(&WhL[(size_t)(16 * j + col) * 128 + 32 * f + koff]);
      v8f z{}; acc[j] = z;
#pragma unroll
      for (int f = 0; f < 4; ++f)
        acc[j] = WMMA_BF16(gA[f], bfr[f], acc[j]);
    }

    // h_new = ff1 + sigmoid(ta+tb) * (ff2 - ff1)
#pragma unroll
    for (int j2 = 0; j2 < 2; ++j2) {
      int u = 16 * j2 + col;
      float b1 = bhL[u], b2_ = bhL[32 + u], b3 = bhL[64 + u], b4 = bhL[96 + u];
#pragma unroll
      for (int v = 0; v < 8; ++v) {
        float f1 = fast_tanh(acc[j2][v] + b1);
        float f2 = fast_tanh(acc[j2 + 2][v] + b2_);
        float ts = (acc[j2 + 4][v] + b3) + (acc[j2 + 6][v] + b4);
        float ti = fast_sigmoid(ts);
        float hn = f1 + ti * (f2 - f1);
        int m = rowOff + v;
        seq[((size_t)(bb0 + m) * 512 + t) * 32 + u] = hn;
        hbuf[m * 32 + u] = (__bf16)hn;
      }
    }
    __syncthreads();
    hA = frag_from_bf16(&hbuf[r * 32 + koff]);
  }
}

// ---- kernel 4: gelu -> W2 head -> max over T -> softmax ---------------------
__global__ void __launch_bounds__(64) k_final(
    const float* __restrict__ seq, const float* __restrict__ W2,
    const float* __restrict__ b2, float* __restrict__ out) {
  __shared__ float wL[96];
  __shared__ float bL[3];
  __shared__ float red[64][3];
  const int b = blockIdx.x, tid = threadIdx.x;
  for (int i = tid; i < 96; i += 64) wL[i] = W2[i];
  if (tid < 3) bL[tid] = b2[tid];
  __syncthreads();

  float m0 = -3.4e38f, m1 = -3.4e38f, m2 = -3.4e38f;
  for (int t = tid; t < 512; t += 64) {
    const float* s = seq + ((size_t)b * 512 + t) * 32;
    float l0 = bL[0], l1 = bL[1], l2 = bL[2];
#pragma unroll
    for (int u = 0; u < 32; ++u) {
      float y = gelu_exact(s[u]);
      l0 += y * wL[u]; l1 += y * wL[32 + u]; l2 += y * wL[64 + u];
    }
    m0 = fmaxf(m0, l0); m1 = fmaxf(m1, l1); m2 = fmaxf(m2, l2);
  }
  red[tid][0] = m0; red[tid][1] = m1; red[tid][2] = m2;
  __syncthreads();
  if (tid == 0) {
    float a0 = red[0][0], a1 = red[0][1], a2 = red[0][2];
    for (int i = 1; i < 64; ++i) {
      a0 = fmaxf(a0, red[i][0]); a1 = fmaxf(a1, red[i][1]); a2 = fmaxf(a2, red[i][2]);
    }
    float mx = fmaxf(a0, fmaxf(a1, a2));
    float e0 = expf(a0 - mx), e1 = expf(a1 - mx), e2 = expf(a2 - mx);
    float inv = 1.0f / (e0 + e1 + e2);
    out[b * 3 + 0] = e0 * inv; out[b * 3 + 1] = e1 * inv; out[b * 3 + 2] = e2 * inv;
  }
}

extern "C" void kernel_launch(void* const* d_in, const int* in_sizes, int n_in,
                              void* d_out, int out_size, void* d_ws, size_t ws_size,
                              hipStream_t stream) {
  (void)in_sizes; (void)n_in; (void)out_size; (void)ws_size;
  const float* x    = (const float*)d_in[0];
  const float* W0   = (const float*)d_in[1];
  const float* b0   = (const float*)d_in[2];
  const float* Wb   = (const float*)d_in[3];
  const float* bb   = (const float*)d_in[4];
  const float* Wff1 = (const float*)d_in[5];
  const float* bff1 = (const float*)d_in[6];
  const float* Wff2 = (const float*)d_in[7];
  const float* bff2 = (const float*)d_in[8];
  const float* Wta  = (const float*)d_in[9];
  const float* bta  = (const float*)d_in[10];
  const float* Wtb  = (const float*)d_in[11];
  const float* btb  = (const float*)d_in[12];
  const float* W2   = (const float*)d_in[13];
  const float* b2   = (const float*)d_in[14];

  char* ws = (char*)d_ws;
  __bf16* xb   = (__bf16*)ws;                           // 67,108,864 * 2 = 128 MiB
  __bf16* W0b  = (__bf16*)(ws + (size_t)134217728);     // 524,288 * 2 = 1 MiB
  __bf16* Wbxb = (__bf16*)(ws + (size_t)135266304);     // 65,536 * 2 = 128 KiB
  __bf16* h0   = (__bf16*)(ws + (size_t)135397376);     // 33,554,432 * 2 = 64 MiB
  float*  pre  = (float*)(ws + (size_t)202506240);      // 8,388,608 * 4 = 32 MiB
  float*  seq  = (float*)(ws + (size_t)236060672);      // 2,097,152 * 4 = 8 MiB

  k_cvt_f32_bf16<<<32768, 256, 0, stream>>>(x, xb, 8388608);
  k_cvt_f32_bf16<<<256, 256, 0, stream>>>(W0, W0b, 65536);
  k_cvt_wbx<<<32, 256, 0, stream>>>(Wb, Wbxb);
  k_gemm1<<<1024, 256, 65536, stream>>>(xb, W0b, b0, h0);
  k_gemm2<<<256, 256, 65536, stream>>>(h0, Wbxb, bb, pre);
  k_scan<<<8, 32, 0, stream>>>(pre, Wb, Wff1, bff1, Wff2, bff2,
                               Wta, bta, Wtb, btb, seq);
  k_final<<<128, 64, 0, stream>>>(seq, W2, b2, (float*)d_out);
}
